// Attention_layer_81612968559057
// MI455X (gfx1250) — matrix operand order repaired and checked by reading, not compiled, not run on hardware
//
#include <hip/hip_runtime.h>
#include <hip/hip_bf16.h>

// ---------- types ----------
typedef __attribute__((ext_vector_type(16))) __bf16 v16bf;
typedef __attribute__((ext_vector_type(8)))  __bf16 v8bf;
typedef __attribute__((ext_vector_type(4)))  __bf16 v4bf;
typedef __attribute__((ext_vector_type(8)))  float  v8f;

#define N_  64
#define L_  196
#define T_  32
#define D_  512

#define LDS_STRIDE 40   // bf16 elems per staged column row (32 data + 8 pad) = 80B, 16B aligned

// ---------------------------------------------------------------------------
// f32 -> bf16 conversion, 4 elements/thread (float4 load, 8B bf16x4 store)
// ---------------------------------------------------------------------------
__global__ void cvt_bf16_kernel(const float* __restrict__ src,
                                __bf16* __restrict__ dst, int n4) {
    int i = blockIdx.x * blockDim.x + threadIdx.x;
    if (i >= n4) return;
    float4 v = ((const float4*)src)[i];
    v4bf o;
    o[0] = (__bf16)v.x; o[1] = (__bf16)v.y; o[2] = (__bf16)v.z; o[3] = (__bf16)v.w;
    ((v4bf*)dst)[i] = o;
}

// ---------------------------------------------------------------------------
// Transpose 512x512 f32 [K,N] -> bf16 [N,K]
// ---------------------------------------------------------------------------
__global__ void wtrans_kernel(const float* __restrict__ W, __bf16* __restrict__ Wt) {
    int idx = blockIdx.x * blockDim.x + threadIdx.x;  // over 512*512
    int n = idx >> 9, k = idx & 511;
    Wt[idx] = (__bf16)W[k * D_ + n];
}

// ---------------------------------------------------------------------------
// GEMV: s[row] = dot(X[row,:512], w). One wave per row, 8 rows / 256-thr block.
// ---------------------------------------------------------------------------
__global__ void gemv_kernel(const float* __restrict__ X, const float* __restrict__ w,
                            float* __restrict__ s, int rows) {
    int lane = threadIdx.x & 31;
    int wave = threadIdx.x >> 5;
    int row  = blockIdx.x * 8 + wave;
    if (row >= rows) return;
    const float* x = X + (size_t)row * D_;
    float acc = 0.f;
    #pragma unroll
    for (int i = lane; i < D_; i += 32) acc += x[i] * w[i];
    #pragma unroll
    for (int off = 16; off; off >>= 1) acc += __shfl_xor(acc, off);
    if (lane == 0) s[row] = acc;
}

// ---------------------------------------------------------------------------
// WMMA A-fragment loader (contiguous k = 16h + s order, wave32)
// A 16x32 (MxK): lanes 0-15 -> K k0..k0+15; lanes 16-31 -> K k0+16..k0+31
// ---------------------------------------------------------------------------
__device__ __forceinline__ v16bf load_a_frag(const __bf16* __restrict__ X, int ld,
                                             int row0, int k0, int lane) {
    const __bf16* p = X + (size_t)(row0 + (lane & 15)) * ld + k0 + ((lane >> 4) << 4);
    v8bf lo = *(const v8bf*)(p);
    v8bf hi = *(const v8bf*)(p + 8);
    v16bf a;
    #pragma unroll
    for (int j = 0; j < 8; ++j) { a[j] = lo[j]; a[j + 8] = hi[j]; }
    return a;
}

// ---------------------------------------------------------------------------
// Fused GLU GEMM: out = tanh(X@W1 + b1) * sigmoid(X@W2 + b2)
// X bf16 [M,512] row-major, W1t/W2t bf16 [512,512] K-major.
// Block = 256 thr = 8 waves (4 row-waves x 2 col-waves); block tile 64x128.
// Weight tiles (128 cols x 32 K, both matrices = 16KB/stage) are double-buffer
// staged into LDS with GLOBAL_LOAD_ASYNC_TO_LDS_B128 (ASYNCcnt), then read as
// B-fragments with 16B ds loads. A-fragments stream direct from global.
// Inner body: issue A global load + all 16 ds loads first, then 8-WMMA burst,
// so LDS latency is covered by one dscnt wait instead of one per WMMA.
// grid = (M/64, 4)
// ---------------------------------------------------------------------------
__global__ void glu_wmma_kernel(const __bf16* __restrict__ X,
                                const __bf16* __restrict__ W1t,
                                const __bf16* __restrict__ W2t,
                                const float* __restrict__ b1,
                                const float* __restrict__ b2,
                                float* __restrict__ out) {
    __shared__ __bf16 bstage[2][2][128][LDS_STRIDE];   // [stage][matrix][col][k] = 40,960 B

    const int lane = threadIdx.x & 31;
    const int wave = threadIdx.x >> 5;
    const int wrow = wave & 3, wcol = wave >> 2;
    const int row0   = blockIdx.x * 64 + wrow * 16;
    const int colblk = blockIdx.y * 128;

    // async-fill assignment: one (matrix, column) pair per thread, 64B per stage
    const int fm = threadIdx.x >> 7;     // 0 -> W1t, 1 -> W2t
    const int fc = threadIdx.x & 127;    // column within block tile
    const __bf16* fsrc = (fm ? W2t : W1t) + (size_t)(colblk + fc) * D_;
    const unsigned ldst0 = (unsigned)(uintptr_t)&bstage[0][fm][fc][0];
    const unsigned ldst1 = (unsigned)(uintptr_t)&bstage[1][fm][fc][0];

    // 64 bytes = 4 x async b128; instruction offset applies to BOTH lds+global addr
    auto fill = [&](unsigned ldst, int k0) {
        unsigned long long g = (unsigned long long)(uintptr_t)(fsrc + k0);
        asm volatile("global_load_async_to_lds_b128 %0, %1, off"           :: "v"(ldst), "v"(g) : "memory");
        asm volatile("global_load_async_to_lds_b128 %0, %1, off offset:16" :: "v"(ldst), "v"(g) : "memory");
        asm volatile("global_load_async_to_lds_b128 %0, %1, off offset:32" :: "v"(ldst), "v"(g) : "memory");
        asm volatile("global_load_async_to_lds_b128 %0, %1, off offset:48" :: "v"(ldst), "v"(g) : "memory");
    };

    v8f acc1[4] = {}; v8f acc2[4] = {};

    fill(ldst0, 0);                                   // prime stage 0
    for (int i = 0; i < 16; ++i) {
        const int k0  = i << 5;
        const int cur = i & 1;
        if (i + 1 < 16) {
            fill(cur ? ldst0 : ldst1, k0 + 32);       // prefetch next stage
            asm volatile("s_wait_asynccnt 0x4" ::: "memory");  // first 4 (in-order) done
        } else {
            asm volatile("s_wait_asynccnt 0x0" ::: "memory");
        }
        __syncthreads();                              // stage `cur` visible to all waves

        // ---- issue phase: A from global + 16 independent ds_load_b128 ----
        v16bf a = load_a_frag(X, D_, row0, k0, lane);

        const int kl = (lane >> 4) << 4;              // lanes 16-31 -> K=16..31
        v16bf frag[8];                                // [0..3]=W1 tiles, [4..7]=W2 tiles
        #pragma unroll
        for (int t = 0; t < 4; ++t) {
            const int nl = wcol * 64 + t * 16 + (lane & 15);
            const __bf16* pA = &bstage[cur][0][nl][kl];
            v8bf alo = *(const v8bf*)pA;
            v8bf ahi = *(const v8bf*)(pA + 8);
            const __bf16* pB = &bstage[cur][1][nl][kl];
            v8bf blo = *(const v8bf*)pB;
            v8bf bhi = *(const v8bf*)(pB + 8);
            #pragma unroll
            for (int j = 0; j < 8; ++j) {
                frag[t][j]     = alo[j]; frag[t][j + 8]     = ahi[j];
                frag[4 + t][j] = blo[j]; frag[4 + t][j + 8] = bhi[j];
            }
        }

        // ---- compute phase: 8 back-to-back WMMAs sharing one A fragment ----
        #pragma unroll
        for (int t = 0; t < 4; ++t) {
            acc1[t] = __builtin_amdgcn_wmma_f32_16x16x32_bf16(
                false, a, false, frag[t], (short)0, acc1[t], false, false);
            acc2[t] = __builtin_amdgcn_wmma_f32_16x16x32_bf16(
                false, a, false, frag[4 + t], (short)0, acc2[t], false, false);
        }
        __syncthreads();                              // all reads done before refill
    }

    // epilogue: C/D layout -> row = row0 + j + 8*(lane>=16), col = col0+16t+(lane&15)
    const int rbase = row0 + 8 * (lane >> 4);
    const int c     = lane & 15;
    #pragma unroll
    for (int t = 0; t < 4; ++t) {
        int col = colblk + wcol * 64 + t * 16 + c;
        float bias1 = b1[col], bias2 = b2[col];
        #pragma unroll
        for (int j = 0; j < 8; ++j) {
            float g = tanhf(acc1[t][j] + bias1);
            float z = acc2[t][j] + bias2;
            float s = 1.0f / (1.0f + __expf(-z));
            out[(size_t)(rbase + j) * D_ + col] = g * s;
        }
    }
}

// ---------------------------------------------------------------------------
// Softmax over L=196: att[n,t,:] = softmax(s2[n,t] + s1[n,:] + b_att)
// One wave per (n,t) row, 8 rows per 256-thr block; grid = 2048/8 = 256.
// ---------------------------------------------------------------------------
__global__ void softmax_kernel(const float* __restrict__ s1, const float* __restrict__ s2,
                               const float* __restrict__ b_att, float* __restrict__ att) {
    int lane = threadIdx.x & 31;
    int wave = threadIdx.x >> 5;
    int row  = blockIdx.x * 8 + wave;      // 0 .. N_*T_-1
    int n    = row / T_;
    float base = s2[row] + b_att[0];
    const float* s1n = s1 + (size_t)n * L_;

    float vals[7]; float m = -3.0e38f;
    #pragma unroll
    for (int i = 0; i < 7; ++i) {
        int l = lane + i * 32;
        float v = (l < L_) ? (s1n[l] + base) : -3.0e38f;
        vals[i] = v; m = fmaxf(m, v);
    }
    #pragma unroll
    for (int off = 16; off; off >>= 1) m = fmaxf(m, __shfl_xor(m, off));
    float sum = 0.f;
    #pragma unroll
    for (int i = 0; i < 7; ++i) {
        int l = lane + i * 32;
        float e = (l < L_) ? __expf(vals[i] - m) : 0.f;
        vals[i] = e; sum += e;
    }
    #pragma unroll
    for (int off = 16; off; off >>= 1) sum += __shfl_xor(sum, off);
    float inv = 1.0f / sum;
    #pragma unroll
    for (int i = 0; i < 7; ++i) {
        int l = lane + i * 32;
        if (l < L_) att[(size_t)row * L_ + l] = vals[i] * inv;
    }
}

// ---------------------------------------------------------------------------
// f_hat: out[n,:,:] += att[n,32x196] @ f1[n,196x512], K padded 196->224 w/ zeros.
// Block = 128 thr = 4 waves; wave -> (rowtile in {0,16}, 64-col group).
// grid = (64, 4): block covers 128 cols. out already holds f2_emb.
// ---------------------------------------------------------------------------
__global__ void fhat_wmma_kernel(const float* __restrict__ att,
                                 const float* __restrict__ f1,
                                 float* __restrict__ out) {
    int n    = blockIdx.x;
    int lane = threadIdx.x & 31;
    int wave = threadIdx.x >> 5;
    int row0 = (wave & 1) * 16;
    int col0 = blockIdx.y * 128 + (wave >> 1) * 64;

    const float* attn = att + (size_t)n * T_ * L_;
    const float* f1n  = f1  + (size_t)n * L_ * D_;

    v8f acc[4] = {};
    for (int k0 = 0; k0 < 224; k0 += 32) {
        // A fragment from f32 att with zero-padding past K=196
        v16bf a;
        {
            int r  = row0 + (lane & 15);
            int kb = k0 + ((lane >> 4) << 4);
            const float* ar = attn + (size_t)r * L_;
            #pragma unroll
            for (int j = 0; j < 8; ++j) {
                int ka = kb + j, kc = kb + 8 + j;
                a[j]     = (ka < L_) ? (__bf16)ar[ka] : (__bf16)0.0f;
                a[j + 8] = (kc < L_) ? (__bf16)ar[kc] : (__bf16)0.0f;
            }
        }
        #pragma unroll
        for (int t = 0; t < 4; ++t) {
            v16bf b;
            int col = col0 + t * 16 + (lane & 15);
            int kb  = k0 + ((lane >> 4) << 4);
            #pragma unroll
            for (int j = 0; j < 16; ++j) {
                int k = kb + j;
                b[j] = (k < L_) ? (__bf16)f1n[(size_t)k * D_ + col] : (__bf16)0.0f;
            }
            acc[t] = __builtin_amdgcn_wmma_f32_16x16x32_bf16(
                false, a, false, b, (short)0, acc[t], false, false);
        }
    }
    int rbase = row0 + 8 * (lane >> 4);
    int c     = lane & 15;
    float* on = out + (size_t)n * T_ * D_;
    #pragma unroll
    for (int t = 0; t < 4; ++t)
        #pragma unroll
        for (int j = 0; j < 8; ++j)
            on[(size_t)(rbase + j) * D_ + col0 + t * 16 + c] += acc[t][j];
}

// ---------------------------------------------------------------------------
// launch
// ---------------------------------------------------------------------------
extern "C" void kernel_launch(void* const* d_in, const int* in_sizes, int n_in,
                              void* d_out, int out_size, void* d_ws, size_t ws_size,
                              hipStream_t stream) {
    const float* f1    = (const float*)d_in[0];   // [N,L,D]
    const float* f2    = (const float*)d_in[1];   // [N,T,D]
    const float* w_att = (const float*)d_in[2];   // [D]
    const float* b_att = (const float*)d_in[3];   // scalar
    const float* n1w1  = (const float*)d_in[4];
    const float* n1b1  = (const float*)d_in[5];
    const float* n1w2  = (const float*)d_in[6];
    const float* n1b2  = (const float*)d_in[7];
    const float* n2w1  = (const float*)d_in[8];
    const float* n2b1  = (const float*)d_in[9];
    const float* n2w2  = (const float*)d_in[10];
    const float* n2b2  = (const float*)d_in[11];

    const int NLD = N_ * L_ * D_;   // 6,422,528
    const int NTD = N_ * T_ * D_;   // 1,048,576
    float* out_f2  = (float*)d_out;                 // f2_emb + f_hat [N,T,D]
    float* out_f1e = (float*)d_out + NTD;           // f1_emb [N,L,D]
    float* out_att = (float*)d_out + NTD + NLD;     // att [N,T,L]

    char* ws = (char*)d_ws;
    __bf16* f1b   = (__bf16*)ws;                           // 12,845,056 B
    __bf16* f2b   = (__bf16*)(ws + 12845056);              //  2,097,152 B
    __bf16* w1t_1 = (__bf16*)(ws + 14942208);              //    524,288 B each
    __bf16* w2t_1 = w1t_1 + 262144;
    __bf16* w1t_2 = w2t_1 + 262144;
    __bf16* w2t_2 = w1t_2 + 262144;
    float*  s1    = (float*)(ws + 17039360);               // 50,176 B
    float*  s2    = (float*)(ws + 17089536);               //  8,192 B

    // 1) features -> bf16
    cvt_bf16_kernel<<<(NLD / 4 + 255) / 256, 256, 0, stream>>>(f1, f1b, NLD / 4);
    cvt_bf16_kernel<<<(NTD / 4 + 255) / 256, 256, 0, stream>>>(f2, f2b, NTD / 4);

    // 2) weights -> bf16, K-major
    wtrans_kernel<<<1024, 256, 0, stream>>>(n1w1, w1t_1);
    wtrans_kernel<<<1024, 256, 0, stream>>>(n1w2, w2t_1);
    wtrans_kernel<<<1024, 256, 0, stream>>>(n2w1, w1t_2);
    wtrans_kernel<<<1024, 256, 0, stream>>>(n2w2, w2t_2);

    // 3) attention scores s1 = f1 @ w_att, s2 = f2 @ w_att
    gemv_kernel<<<(N_ * L_) / 8, 256, 0, stream>>>(f1, w_att, s1, N_ * L_);
    gemv_kernel<<<(N_ * T_) / 8, 256, 0, stream>>>(f2, w_att, s2, N_ * T_);

    // 4) GLU paths (WMMA bf16, f32 accumulate, async-LDS staged weights)
    glu_wmma_kernel<<<dim3((N_ * L_) / 64, 4), 256, 0, stream>>>(
        f1b, w1t_1, w2t_1, n1b1, n1b2, out_f1e);
    glu_wmma_kernel<<<dim3((N_ * T_) / 64, 4), 256, 0, stream>>>(
        f2b, w1t_2, w2t_2, n2b1, n2b2, out_f2);

    // 5) softmax attention
    softmax_kernel<<<(N_ * T_) / 8, 256, 0, stream>>>(s1, s2, b_att, out_att);

    // 6) f_hat accumulate into out_f2 (WMMA, zero-padded K tail)
    fhat_wmma_kernel<<<dim3(N_, 4), 128, 0, stream>>>(out_att, f1, out_f2);
}
